// MambaBlock_68247030333785
// MI455X (gfx1250) — matrix operand order repaired and checked by reading, not compiled, not run on hardware
//
#include <hip/hip_runtime.h>

// ---------------- problem constants ----------------
#define MROWS 32768        // B_SZ * SEQ_L
#define K1    768          // D_MODEL
#define DIN   1536         // D_INNER
#define NP1   1664         // padded N for gemm1 (13 * 128), covers 1544 live cols
#define NREAL 1544         // 1536 xc cols + 8 z cols
#define NN    8            // N = D_STATE/2
#define CH    128          // scan chunk length
#define NCHK  32           // 4096 / CH
#define SEQL  4096
#define BSZ   8
#define NK1   (K1 / 32)    // K-tiles for gemm1

typedef __attribute__((ext_vector_type(16))) __bf16 v16bf;
typedef __attribute__((ext_vector_type(8)))  __bf16 v8bf;
typedef __attribute__((ext_vector_type(8)))  float  v8f;

union Frag16 { v16bf v; v8bf h[2]; };
union BF4    { __bf16 b[4]; uint2 u; };

static __device__ inline v8f wmma_bf16(v16bf a, v16bf b, v8f c) {
  // (neg_a, A, neg_b, B, c_mod, C, reuse_a, reuse_b)
  return __builtin_amdgcn_wmma_f32_16x16x32_bf16(false, a, false, b, (short)0, c,
                                                 false, false);
}

// ---- CDNA5 async global->LDS copy (16B per lane), tracked by ASYNCcnt ----
static __device__ inline void async_copy16(unsigned lds_addr, const void* gaddr) {
  asm volatile("global_load_async_to_lds_b128 %0, %1, off"
               :: "v"(lds_addr), "v"(gaddr) : "memory");
}
static __device__ inline void wait_async0() {
  asm volatile("s_wait_asynccnt 0x0" ::: "memory");
}
static __device__ inline unsigned lds_off(const void* p) {
  // low 32 bits of a generic pointer to __shared__ == LDS byte offset
  return (unsigned)(unsigned long long)p;
}

// ---------------- prep kernels ----------------
__global__ void prep_a_kernel(const float* __restrict__ x, __bf16* __restrict__ xA,
                              long n4) {
  long idx = (long)blockIdx.x * blockDim.x + threadIdx.x;
  if (idx >= n4) return;
  float4 v = ((const float4*)x)[idx];
  BF4 t;
  t.b[0] = (__bf16)v.x; t.b[1] = (__bf16)v.y;
  t.b[2] = (__bf16)v.z; t.b[3] = (__bf16)v.w;
  ((uint2*)xA)[idx] = t.u;
}

// Bmat[n][k] = W_in[k][n] for n < 1544 (N-major, K contiguous), zero-padded to NP1.
__global__ void prep_b_kernel(const float* __restrict__ W_in, __bf16* __restrict__ Bmat) {
  long idx = (long)blockIdx.x * blockDim.x + threadIdx.x;
  if (idx >= (long)NP1 * K1) return;
  int k = (int)(idx % K1);
  int n = (int)(idx / K1);
  float v = (n < NREAL) ? W_in[(long)k * 3072 + n] : 0.0f;
  Bmat[idx] = (__bf16)v;
}

// W2[j][k], j<8: W_dt[:, j]; 8..15: W_x[:,16+j-8] (B); 16..23: W_x[:,24+j-16] (C); else 0
__global__ void prep_w2_kernel(const float* __restrict__ W_dt, const float* __restrict__ W_x,
                               const float* __restrict__ b_dt, const float* __restrict__ b_x,
                               __bf16* __restrict__ W2, float* __restrict__ bias2) {
  long idx = (long)blockIdx.x * blockDim.x + threadIdx.x;
  if (idx >= 32L * DIN) return;
  int j = (int)(idx / DIN);
  int k = (int)(idx % DIN);
  float v = 0.0f;
  if (j < 8)       v = W_dt[(long)k * DIN + j];
  else if (j < 16) v = W_x[(long)k * 32 + 16 + (j - 8)];
  else if (j < 24) v = W_x[(long)k * 32 + 24 + (j - 16)];
  W2[idx] = (__bf16)v;
  if (idx < 32) {
    int jj = (int)idx;
    float bv = 0.0f;
    if (jj < 8)       bv = b_dt[jj];
    else if (jj < 16) bv = b_x[16 + (jj - 8)];
    else if (jj < 24) bv = b_x[24 + (jj - 16)];
    bias2[jj] = bv;
  }
}

// ---------------- GEMM 1: xz = x @ W_in[:, 0:1544] + b_in ----------------
// Double-buffered LDS pipeline fed by async global->LDS copies (ASYNCcnt),
// consumed by v_wmma_f32_16x16x32_bf16.
__global__ __launch_bounds__(256) void gemm1_kernel(
    const __bf16* __restrict__ A, const __bf16* __restrict__ Bw,
    const float* __restrict__ b_in, float* __restrict__ xc, float* __restrict__ z8) {
  __shared__ __attribute__((aligned(16))) __bf16 As[2][128 * 40];
  __shared__ __attribute__((aligned(16))) __bf16 Bs[2][128 * 40];
  const int tid   = threadIdx.x;
  const int mBase = blockIdx.y * 128;
  const int nBase = blockIdx.x * 128;
  const int w    = tid >> 5, lane = tid & 31;
  const int half = lane >> 4, l16 = lane & 15;
  const int waveM = (w & 3) * 32, waveN = (w >> 2) * 64;

  // per-thread staging coordinates: 2 groups of 8 bf16 for A, 2 for B
  const int r0 = tid >> 2,          s0 = (tid & 3) * 8;
  const int r1 = (tid + 256) >> 2,  s1 = ((tid + 256) & 3) * 8;

  const __bf16* gA0 = A  + (size_t)(mBase + r0) * K1 + s0;
  const __bf16* gA1 = A  + (size_t)(mBase + r1) * K1 + s1;
  const __bf16* gB0 = Bw + (size_t)(nBase + r0) * K1 + s0;
  const __bf16* gB1 = Bw + (size_t)(nBase + r1) * K1 + s1;

  unsigned lA0[2], lA1[2], lB0[2], lB1[2];
#pragma unroll
  for (int p = 0; p < 2; ++p) {
    lA0[p] = lds_off(&As[p][r0 * 40 + s0]);
    lA1[p] = lds_off(&As[p][r1 * 40 + s1]);
    lB0[p] = lds_off(&Bs[p][r0 * 40 + s0]);
    lB1[p] = lds_off(&Bs[p][r1 * 40 + s1]);
  }

  // prologue: prefetch K-tile 0 into buffer 0
  async_copy16(lA0[0], gA0);
  async_copy16(lA1[0], gA1);
  async_copy16(lB0[0], gB0);
  async_copy16(lB1[0], gB1);

  v8f acc[2][4];
#pragma unroll
  for (int mi = 0; mi < 2; ++mi)
#pragma unroll
    for (int ni = 0; ni < 4; ++ni)
      acc[mi][ni] = (v8f){0.f, 0.f, 0.f, 0.f, 0.f, 0.f, 0.f, 0.f};

  for (int kk = 0; kk < NK1; ++kk) {
    const int cur = kk & 1, nxt = cur ^ 1;
    wait_async0();        // my async writes to buf[cur] landed
    __syncthreads();      // everyone's landed; everyone's reads of buf[nxt] done

    if (kk + 1 < NK1) {   // fire-and-forget next tile while we compute
      const int ko = (kk + 1) * 32;
      async_copy16(lA0[nxt], gA0 + ko);
      async_copy16(lA1[nxt], gA1 + ko);
      async_copy16(lB0[nxt], gB0 + ko);
      async_copy16(lB1[nxt], gB1 + ko);
    }

    Frag16 af[2], bfg[4];
    const int kbA = half * 8;
#pragma unroll
    for (int mi = 0; mi < 2; ++mi) {
      int row = waveM + mi * 16 + l16;
      af[mi].h[0] = *(const v8bf*)&As[cur][row * 40 + kbA];
      af[mi].h[1] = *(const v8bf*)&As[cur][row * 40 + kbA + 16];
    }
    const int kbB = half * 8;
#pragma unroll
    for (int ni = 0; ni < 4; ++ni) {
      int col = waveN + ni * 16 + l16;
      bfg[ni].h[0] = *(const v8bf*)&Bs[cur][col * 40 + kbB];
      bfg[ni].h[1] = *(const v8bf*)&Bs[cur][col * 40 + kbB + 16];
    }
#pragma unroll
    for (int mi = 0; mi < 2; ++mi)
#pragma unroll
      for (int ni = 0; ni < 4; ++ni)
        acc[mi][ni] = wmma_bf16(af[mi].v, bfg[ni].v, acc[mi][ni]);
  }

  // epilogue: add bias, split into xc / z8
#pragma unroll
  for (int mi = 0; mi < 2; ++mi)
#pragma unroll
    for (int ni = 0; ni < 4; ++ni)
#pragma unroll
      for (int r = 0; r < 8; ++r) {
        int gm = mBase + waveM + mi * 16 + r + half * 8;
        int gn = nBase + waveN + ni * 16 + l16;
        if (gn < NREAL) {
          float v = acc[mi][ni][r] + b_in[gn];
          if (gn < DIN) xc[(size_t)gm * DIN + gn] = v;
          else          z8[(size_t)gm * 8 + (gn - DIN)] = v;
        }
      }
}

// ---------------- depthwise causal conv (k=4) + SiLU ----------------
__global__ void conv_silu_kernel(const float* __restrict__ xc,
                                 const float* __restrict__ conv_w,
                                 const float* __restrict__ conv_b,
                                 __bf16* __restrict__ xcs, float* __restrict__ xn) {
  long idx = (long)blockIdx.x * blockDim.x + threadIdx.x;
  if (idx >= (long)MROWS * DIN) return;
  int c = (int)(idx % DIN);
  long r = idx / DIN;
  int l = (int)(r & (SEQL - 1));
  float acc = conv_b[c];
#pragma unroll
  for (int k = 0; k < 4; ++k) {
    int ls = l - 3 + k;
    if (ls >= 0) acc += conv_w[c * 4 + k] * xc[(r - 3 + k) * DIN + c];
  }
  float s = acc / (1.0f + __expf(-acc));  // silu
  xcs[idx] = (__bf16)s;
  if (c < NN) xn[r * 8 + c] = s;
}

// ---------------- GEMM 2: [decay | B | C] = f(xcs @ W2 + bias2) ----------------
__global__ __launch_bounds__(256) void gemm2_kernel(
    const __bf16* __restrict__ A, const __bf16* __restrict__ Bw,
    const float* __restrict__ bias2, float* __restrict__ decayO,
    float* __restrict__ BmO, float* __restrict__ CmO) {
  __shared__ __attribute__((aligned(16))) __bf16 As[128 * 40];
  __shared__ __attribute__((aligned(16))) __bf16 Bs[32 * 40];
  const int tid   = threadIdx.x;
  const int mBase = blockIdx.x * 128;
  const int w    = tid >> 5, lane = tid & 31;
  const int half = lane >> 4, l16 = lane & 15;
  const int waveM = (w & 3) * 32, waveN = (w >> 2) * 16;

  v8f acc[2];
  acc[0] = (v8f){0.f, 0.f, 0.f, 0.f, 0.f, 0.f, 0.f, 0.f};
  acc[1] = acc[0];

  for (int kk = 0; kk < DIN / 32; ++kk) {
    const int k0 = kk * 32;
    __syncthreads();
#pragma unroll
    for (int it = 0; it < 2; ++it) {
      int g = tid + it * 256;
      int row = g >> 2, seg = g & 3;
      *(uint4*)&As[row * 40 + seg * 8] =
          *(const uint4*)&A[(size_t)(mBase + row) * DIN + k0 + seg * 8];
    }
    if (tid < 128) {
      int row = tid >> 2, seg = tid & 3;
      *(uint4*)&Bs[row * 40 + seg * 8] =
          *(const uint4*)&Bw[(size_t)row * DIN + k0 + seg * 8];
    }
    __syncthreads();

    Frag16 af[2], bfg;
    const int kbA = half * 8;
#pragma unroll
    for (int mi = 0; mi < 2; ++mi) {
      int row = waveM + mi * 16 + l16;
      af[mi].h[0] = *(const v8bf*)&As[row * 40 + kbA];
      af[mi].h[1] = *(const v8bf*)&As[row * 40 + kbA + 16];
    }
    const int kbB = half * 8;
    {
      int col = waveN + l16;
      bfg.h[0] = *(const v8bf*)&Bs[col * 40 + kbB];
      bfg.h[1] = *(const v8bf*)&Bs[col * 40 + kbB + 16];
    }
#pragma unroll
    for (int mi = 0; mi < 2; ++mi)
      acc[mi] = wmma_bf16(af[mi].v, bfg.v, acc[mi]);
  }

#pragma unroll
  for (int mi = 0; mi < 2; ++mi)
#pragma unroll
    for (int r = 0; r < 8; ++r) {
      int gm = mBase + waveM + mi * 16 + r + half * 8;
      int gn = waveN + l16;                       // 0..31
      float u = acc[mi][r] + bias2[gn];
      if (gn < 8) {
        // decay = exp(-softplus(u)) = 1/(1+e^u)
        decayO[(size_t)gm * 8 + gn] = 1.0f / (1.0f + __expf(u));
      } else if (gn < 16) {
        BmO[(size_t)gm * 8 + (gn - 8)] = u;
      } else if (gn < 24) {
        CmO[(size_t)gm * 8 + (gn - 16)] = u;
      }
    }
}

// ---------------- chunked linear-recurrence scan ----------------
__global__ __launch_bounds__(64) void scan_phase1_kernel(
    const float* __restrict__ decay, const float* __restrict__ Bm,
    const float* __restrict__ xn, float* __restrict__ S, float* __restrict__ P) {
  const int chunk = blockIdx.x, b = blockIdx.y;
  __shared__ float ds[CH * 8], bs[CH * 8], xs[CH * 8];
  const int t = threadIdx.x;
  const size_t base = ((size_t)b * SEQL + (size_t)chunk * CH) * 8;
  for (int q = t; q < CH * 8; q += 64) {
    ds[q] = decay[base + q];
    bs[q] = Bm[base + q];
    xs[q] = xn[base + q];
  }
  __syncthreads();
  const int i = t >> 3, j = t & 7;
  float h = 0.0f, p = 1.0f;
  for (int s = 0; s < CH; ++s) {
    float d = ds[s * 8 + j];
    h = h * d + bs[s * 8 + j] * xs[s * 8 + i];
    p *= d;
  }
  const int cidx = b * NCHK + chunk;
  S[((size_t)cidx * 8 + i) * 8 + j] = h;
  if (i == 0) P[(size_t)cidx * 8 + j] = p;
}

__global__ __launch_bounds__(512) void scan_carry_kernel(
    const float* __restrict__ S, const float* __restrict__ P, float* __restrict__ hin) {
  const int t = threadIdx.x;  // 512 = 8b * 8i * 8j
  const int b = t >> 6, i = (t >> 3) & 7, j = t & 7;
  float h = 0.0f;
  for (int c = 0; c < NCHK; ++c) {
    const int cidx = b * NCHK + c;
    hin[((size_t)cidx * 8 + i) * 8 + j] = h;
    h = P[(size_t)cidx * 8 + j] * h + S[((size_t)cidx * 8 + i) * 8 + j];
  }
}

__global__ __launch_bounds__(64) void scan_phase2_kernel(
    const float* __restrict__ decay, const float* __restrict__ Bm,
    const float* __restrict__ Cm, const float* __restrict__ xn,
    const float* __restrict__ hin, float* __restrict__ ys) {
  const int chunk = blockIdx.x, b = blockIdx.y;
  __shared__ float ds[CH * 8], bs[CH * 8], cs[CH * 8], xs[CH * 8];
  const int t = threadIdx.x;
  const size_t base = ((size_t)b * SEQL + (size_t)chunk * CH) * 8;
  for (int q = t; q < CH * 8; q += 64) {
    ds[q] = decay[base + q];
    bs[q] = Bm[base + q];
    cs[q] = Cm[base + q];
    xs[q] = xn[base + q];
  }
  __syncthreads();
  const int i = t >> 3, j = t & 7;
  const int cidx = b * NCHK + chunk;
  float h = hin[((size_t)cidx * 8 + i) * 8 + j];
  const size_t row0 = (size_t)b * SEQL + (size_t)chunk * CH;
  for (int s = 0; s < CH; ++s) {
    float d = ds[s * 8 + j];
    h = h * d + bs[s * 8 + j] * xs[s * 8 + i];
    float v = cs[s * 8 + j] * h;
    v += __shfl_xor(v, 1, 8);
    v += __shfl_xor(v, 2, 8);
    v += __shfl_xor(v, 4, 8);
    if (j == 0) ys[(row0 + s) * 8 + i] = v;
  }
}

// ---------------- final: out = (ys * silu(z8)) @ W_out[:8,:] + b_out ----------------
__global__ __launch_bounds__(256) void final_out_kernel(
    const float* __restrict__ ys, const float* __restrict__ z8,
    const float* __restrict__ W_out, const float* __restrict__ b_out,
    float* __restrict__ out) {
  const int r = blockIdx.y;
  const int m = blockIdx.x * 256 + threadIdx.x;  // < 768
  float acc = b_out[m];
#pragma unroll
  for (int i = 0; i < 8; ++i) {
    float z = z8[(size_t)r * 8 + i];
    float g = ys[(size_t)r * 8 + i] * (z / (1.0f + __expf(-z)));
    acc += g * W_out[(size_t)i * K1 + m];
  }
  out[(size_t)r * K1 + m] = acc;
}

// ---------------- launch ----------------
extern "C" void kernel_launch(void* const* d_in, const int* in_sizes, int n_in,
                              void* d_out, int out_size, void* d_ws, size_t ws_size,
                              hipStream_t stream) {
  const float* x      = (const float*)d_in[0];
  const float* W_in   = (const float*)d_in[1];
  const float* b_in   = (const float*)d_in[2];
  const float* conv_w = (const float*)d_in[3];
  const float* conv_b = (const float*)d_in[4];
  const float* W_x    = (const float*)d_in[5];
  const float* b_x    = (const float*)d_in[6];
  const float* W_dt   = (const float*)d_in[7];
  const float* b_dt   = (const float*)d_in[8];
  const float* W_out  = (const float*)d_in[9];
  const float* b_out  = (const float*)d_in[10];
  float* out = (float*)d_out;

  char* ws = (char*)d_ws;
  size_t off = 0;
  auto alloc = [&](size_t bytes) -> void* {
    off = (off + 255) & ~(size_t)255;
    void* p = ws + off;
    off += bytes;
    return p;
  };

  __bf16* xA    = (__bf16*)alloc((size_t)MROWS * K1 * 2);
  __bf16* Bmat  = (__bf16*)alloc((size_t)NP1 * K1 * 2);
  float*  xc    = (float*) alloc((size_t)MROWS * DIN * 4);
  __bf16* xcs   = (__bf16*)alloc((size_t)MROWS * DIN * 2);
  float*  xn    = (float*) alloc((size_t)MROWS * 8 * 4);
  float*  z8    = (float*) alloc((size_t)MROWS * 8 * 4);
  __bf16* W2    = (__bf16*)alloc((size_t)32 * DIN * 2);
  float*  bias2 = (float*) alloc(32 * 4);
  float*  dec   = (float*) alloc((size_t)MROWS * 8 * 4);
  float*  Bmv   = (float*) alloc((size_t)MROWS * 8 * 4);
  float*  Cmv   = (float*) alloc((size_t)MROWS * 8 * 4);
  float*  Sb    = (float*) alloc((size_t)BSZ * NCHK * 8 * 8 * 4);
  float*  Pb    = (float*) alloc((size_t)BSZ * NCHK * 8 * 4);
  float*  hin   = (float*) alloc((size_t)BSZ * NCHK * 8 * 8 * 4);
  float*  ysb   = (float*) alloc((size_t)MROWS * 8 * 4);

  // 1) bf16 conversions
  {
    long n4 = (long)MROWS * K1 / 4;
    prep_a_kernel<<<(unsigned)((n4 + 255) / 256), 256, 0, stream>>>(x, xA, n4);
    long nb = (long)NP1 * K1;
    prep_b_kernel<<<(unsigned)((nb + 255) / 256), 256, 0, stream>>>(W_in, Bmat);
    long nw = 32L * DIN;
    prep_w2_kernel<<<(unsigned)((nw + 255) / 256), 256, 0, stream>>>(
        W_dt, W_x, b_dt, b_x, W2, bias2);
  }

  // 2) big GEMM (WMMA bf16, async double-buffered LDS): xc + z8
  gemm1_kernel<<<dim3(NP1 / 128, MROWS / 128), 256, 0, stream>>>(xA, Bmat, b_in, xc, z8);

  // 3) depthwise conv + silu
  {
    long n = (long)MROWS * DIN;
    conv_silu_kernel<<<(unsigned)((n + 255) / 256), 256, 0, stream>>>(
        xc, conv_w, conv_b, xcs, xn);
  }

  // 4) skinny GEMM (WMMA bf16): decay / B / C
  gemm2_kernel<<<MROWS / 128, 256, 0, stream>>>(xcs, W2, bias2, dec, Bmv, Cmv);

  // 5) chunked parallel scan
  scan_phase1_kernel<<<dim3(NCHK, BSZ), 64, 0, stream>>>(dec, Bmv, xn, Sb, Pb);
  scan_carry_kernel<<<1, 512, 0, stream>>>(Sb, Pb, hin);
  scan_phase2_kernel<<<dim3(NCHK, BSZ), 64, 0, stream>>>(dec, Bmv, Cmv, xn, hin, ysb);

  // 6) rank-8 output GEMM
  final_out_kernel<<<dim3(K1 / 256, MROWS), 256, 0, stream>>>(ysb, z8, W_out, b_out, out);
}
